// GATv2Encoder_86577950753381
// MI455X (gfx1250) — matrix operand order repaired and checked by reading, not compiled, not run on hardware
//
#include <hip/hip_runtime.h>
#include <hip/hip_bf16.h>
#include <math.h>

// ---------------------------------------------------------------------------
// GATv2 encoder (2 layers) for MI455X / gfx1250.
//   GEMMs        : bf16 inputs, f32 accum via v_wmma_f32_16x16x32_bf16.
//                  A fragments via 2x b128 contiguous loads; B (weights)
//                  pre-packed into WMMA fragment order -> 2x b128 loads.
//                  4 output tiles per wave (64 cols) to amortize A loads.
//   Edge phase   : fp32, segment softmax with ordered-uint atomic max,
//                  atomicAdd segment sums, scatter aggregation.
// ---------------------------------------------------------------------------

#define NODES  25000
#define EDGES  400000
#define FIN    128
#define FEDGE  64
#define HEADS  4
#define CH     32
#define HC     128           // HEADS*CH
#define NEG_SLOPE 0.2f

typedef __attribute__((ext_vector_type(16))) __bf16 v16bf;
typedef __attribute__((ext_vector_type(8)))  __bf16 v8bf;
typedef __attribute__((ext_vector_type(8)))  float  v8f;

// ------------------------- small utility kernels ---------------------------

__global__ void fill_f32(float* p, float v, size_t n) {
    size_t i = (size_t)blockIdx.x * blockDim.x + threadIdx.x;
    if (i < n) p[i] = v;
}

__global__ void fill_u32(unsigned* p, unsigned v, size_t n) {
    size_t i = (size_t)blockIdx.x * blockDim.x + threadIdx.x;
    if (i < n) p[i] = v;
}

__global__ void cvt_f32_to_bf16(const float* __restrict__ in, __bf16* __restrict__ out, size_t n) {
    size_t i = (size_t)blockIdx.x * blockDim.x + threadIdx.x;
    if (i < n) out[i] = (__bf16)in[i];
}

// Pack a row-major f32 weight matrix B[K x Nn] into bf16 WMMA-B fragment
// order: Bp[((kt*Nn + col)*2 + half)*16 + i] = B[(kt*32 + half*16 + i)*Nn + col]
// so each lane's 16 B elements are 32 contiguous bytes (two b128 loads).
__global__ void pack_B_bf16(const float* __restrict__ B, __bf16* __restrict__ Bp,
                            int K, int Nn) {
    size_t o = (size_t)blockIdx.x * blockDim.x + threadIdx.x;
    if (o >= (size_t)K * Nn) return;
    int i    = (int)(o & 15);
    int half = (int)((o >> 4) & 1);
    size_t rest = o >> 5;
    int col = (int)(rest % Nn);
    int kt  = (int)(rest / Nn);
    int k   = kt * 32 + half * 16 + i;
    Bp[o] = (__bf16)B[(size_t)k * Nn + col];
}

// Monotonic float <-> uint mapping so atomicMax(uint) implements float max.
__device__ __forceinline__ unsigned f2ord(float f) {
    unsigned u = __float_as_uint(f);
    return (u >> 31) ? ~u : (u | 0x80000000u);
}
__device__ __forceinline__ float ord2f(unsigned u) {
    return (u >> 31) ? __uint_as_float(u & 0x7fffffffu) : __uint_as_float(~u);
}
#define ORD_NEG_INF 0x007FFFFFu   // f2ord(-inf)

// ----------------------------- WMMA GEMM -----------------------------------
// C[M x Nn] = A[M x K](bf16 row-major) * B[K x Nn](pre-packed bf16) + bias.
// One wave computes a 16x64 strip (4 16x16 tiles); K stepped by 32 per WMMA.
// Fragment layouts used here (wave32; A/B use contiguous K, C/D is the ISA layout):
//   A 16x32 : lanes 0-15 rows M=0..15, per lane K = halfSel*16 + 0..15
//             -> two contiguous 16B chunks.
//   B 32x16 : lane (&15) = column, element i -> K = halfSel*16 + i (packed).
//   C/D     : VGPR i -> row M=i (lanes 0-15) / M=i+8 (lanes 16-31), lane=col.

template <int K>
__global__ __launch_bounds__(256) void gemm_wmma_bf16(
    const __bf16* __restrict__ A, const __bf16* __restrict__ Bp,
    const float* __restrict__ bias, float* __restrict__ C,
    int M, int Nn)
{
    const int wave = threadIdx.x >> 5;
    const int lane = threadIdx.x & 31;
    const int colGroups = Nn >> 6;              // 64 columns per wave
    const int job = blockIdx.x * 8 + wave;
    const int rowTiles = (M + 15) >> 4;
    if (job >= rowTiles * colGroups) return;

    const int tm = job / colGroups;
    const int tg = job - tm * colGroups;

    const int halfSel = lane >> 4;              // 0: lanes 0-15, 1: lanes 16-31
    const int lrow    = lane & 15;

    int arow = tm * 16 + lrow;
    if (arow >= M) arow = M - 1;                // clamp: loads valid, EXEC all-1
    const int bcol0 = tg * 64 + lrow;

    v8f acc[4];
    #pragma unroll
    for (int j = 0; j < 4; ++j) acc[j] = (v8f){0.f,0.f,0.f,0.f,0.f,0.f,0.f,0.f};

    const __bf16* arowp = A + (size_t)arow * K + halfSel * 16;

    #pragma unroll
    for (int k0 = 0; k0 < K; k0 += 32) {
        const __bf16* abase = arowp + k0;
        __builtin_prefetch(abase + 64, 0, 0);   // next K chunk (hint only)
        v8bf a0 = *(const v8bf*)(abase);
        v8bf a1 = *(const v8bf*)(abase + 8);
        v16bf a = __builtin_shufflevector(a0, a1, 0,1,2,3,4,5,6,7,
                                                  8,9,10,11,12,13,14,15);
        const int kt = k0 >> 5;
        #pragma unroll
        for (int j = 0; j < 4; ++j) {
            const __bf16* bp =
                Bp + ((((size_t)kt * Nn + bcol0 + j * 16) * 2 + halfSel) << 4);
            v8bf b0 = *(const v8bf*)(bp);
            v8bf b1 = *(const v8bf*)(bp + 8);
            v16bf b = __builtin_shufflevector(b0, b1, 0,1,2,3,4,5,6,7,
                                                      8,9,10,11,12,13,14,15);
            acc[j] = __builtin_amdgcn_wmma_f32_16x16x32_bf16(
                false, a, false, b, (short)0, acc[j], false, false);
        }
    }

    #pragma unroll
    for (int j = 0; j < 4; ++j) {
        const int col = bcol0 + j * 16;
        const float bv = bias ? bias[col] : 0.f;
        #pragma unroll
        for (int i = 0; i < 8; ++i) {
            int row = tm * 16 + i + halfSel * 8;
            if (row < M) C[(size_t)row * Nn + col] = acc[j][i] + bv;
        }
    }
}

// --------------------------- edge-phase kernels ----------------------------
// 128 threads per edge; each warp (32 lanes) handles exactly one head.

__global__ __launch_bounds__(256) void edge_alpha_kernel(
    const float* __restrict__ xl, const float* __restrict__ xr,
    const float* __restrict__ ef,
    const long long* __restrict__ src, const long long* __restrict__ dst,
    const float* __restrict__ att,
    float* __restrict__ alpha, unsigned* __restrict__ amax_u)
{
    const int t = threadIdx.x;
    const long long e = (long long)blockIdx.x * 2 + (t >> 7);
    if (e >= EDGES) return;
    const int lt = t & 127;
    const int hd = lt >> 5;
    const int c  = lt & 31;

    const long long s = src[e];
    const long long d = dst[e];
    float v = xl[(size_t)s * HC + lt] + xr[(size_t)d * HC + lt]
            + ef[(size_t)e * HC + lt];
    float g = (v > 0.f) ? v : NEG_SLOPE * v;
    float p = g * att[hd * CH + c];
    #pragma unroll
    for (int m = 16; m > 0; m >>= 1) p += __shfl_xor(p, m, 32);
    if (c == 0) {
        alpha[(size_t)e * HEADS + hd] = p;
        atomicMax(&amax_u[(size_t)d * HEADS + hd], f2ord(p));
    }
}

__global__ void amax_fix_kernel(const unsigned* __restrict__ amax_u,
                                float* __restrict__ amax_f, size_t n)
{
    size_t i = (size_t)blockIdx.x * blockDim.x + threadIdx.x;
    if (i >= n) return;
    float f = ord2f(amax_u[i]);
    unsigned bits = __float_as_uint(f);
    if ((bits & 0x7f800000u) == 0x7f800000u) f = 0.f;  // inf/nan -> 0 (ref)
    amax_f[i] = f;
}

__global__ void edge_exp_kernel(float* __restrict__ alpha,
                                const long long* __restrict__ dst,
                                const float* __restrict__ amax_f,
                                float* __restrict__ denom)
{
    size_t i = (size_t)blockIdx.x * blockDim.x + threadIdx.x;
    if (i >= (size_t)EDGES * HEADS) return;
    size_t e = i >> 2;
    int hd   = (int)(i & 3);
    long long d = dst[e];
    float ex = __expf(alpha[i] - amax_f[(size_t)d * HEADS + hd]);
    alpha[i] = ex;
    atomicAdd(&denom[(size_t)d * HEADS + hd], ex);
}

__global__ __launch_bounds__(256) void edge_aggr_kernel(
    const float* __restrict__ xl,
    const long long* __restrict__ src, const long long* __restrict__ dst,
    const float* __restrict__ alpha, const float* __restrict__ denom,
    float* __restrict__ outacc)
{
    const int t = threadIdx.x;
    const long long e = (long long)blockIdx.x * 2 + (t >> 7);
    if (e >= EDGES) return;
    const int lt = t & 127;
    const int hd = lt >> 5;

    const long long s = src[e];
    const long long d = dst[e];
    float a = alpha[(size_t)e * HEADS + hd]
            / (denom[(size_t)d * HEADS + hd] + 1e-16f);
    atomicAdd(&outacc[(size_t)d * HC + lt], xl[(size_t)s * HC + lt] * a);
}

// ---------------------------- node epilogues -------------------------------

__global__ void node_finish1_kernel(const float* __restrict__ acc,
                                    const float* __restrict__ bias,
                                    __bf16* __restrict__ hbf)
{
    size_t i = (size_t)blockIdx.x * blockDim.x + threadIdx.x;
    if (i >= (size_t)NODES * HC) return;
    float v = acc[i] + bias[i & (HC - 1)];
    float h = (v > 0.f) ? v : (__expf(v) - 1.f);   // ELU
    hbf[i] = (__bf16)h;
}

__global__ void node_finish2_kernel(const float* __restrict__ acc,
                                    const float* __restrict__ bias2,
                                    float* __restrict__ out)
{
    size_t i = (size_t)blockIdx.x * blockDim.x + threadIdx.x;
    if (i >= (size_t)NODES * CH) return;
    size_t n = i >> 5;
    int c = (int)(i & 31);
    const float* p = acc + n * HC;
    out[i] = 0.25f * (p[c] + p[CH + c] + p[2 * CH + c] + p[3 * CH + c]) + bias2[c];
}

// ------------------------------- launcher ----------------------------------

extern "C" void kernel_launch(void* const* d_in, const int* in_sizes, int n_in,
                              void* d_out, int out_size, void* d_ws, size_t ws_size,
                              hipStream_t stream)
{
    (void)in_sizes; (void)n_in; (void)out_size; (void)ws_size;

    const float*      x     = (const float*)d_in[0];
    const long long*  ei    = (const long long*)d_in[1];
    const float*      ea    = (const float*)d_in[2];
    const float*      Wl1   = (const float*)d_in[3];
    const float*      bl1   = (const float*)d_in[4];
    const float*      Wr1   = (const float*)d_in[5];
    const float*      br1   = (const float*)d_in[6];
    const float*      We1   = (const float*)d_in[7];
    const float*      att1  = (const float*)d_in[8];
    const float*      bias1 = (const float*)d_in[9];
    const float*      Wl2   = (const float*)d_in[10];
    const float*      bl2   = (const float*)d_in[11];
    const float*      Wr2   = (const float*)d_in[12];
    const float*      br2   = (const float*)d_in[13];
    const float*      We2   = (const float*)d_in[14];
    const float*      att2  = (const float*)d_in[15];
    const float*      bias2 = (const float*)d_in[16];

    const long long* srcI = ei;
    const long long* dstI = ei + EDGES;

    // ---- workspace carve-up ----
    char* ws = (char*)d_ws;
    size_t off = 0;
    auto carve = [&](size_t bytes) -> char* {
        char* p = ws + off;
        off += (bytes + 255) & ~(size_t)255;
        return p;
    };
    __bf16* wl1b = (__bf16*)carve((size_t)HC * HC * 2);      // packed
    __bf16* wr1b = (__bf16*)carve((size_t)HC * HC * 2);      // packed
    __bf16* wl2b = (__bf16*)carve((size_t)HC * HC * 2);      // packed
    __bf16* wr2b = (__bf16*)carve((size_t)HC * HC * 2);      // packed
    __bf16* we1b = (__bf16*)carve((size_t)FEDGE * HC * 2);   // packed
    __bf16* we2b = (__bf16*)carve((size_t)FEDGE * HC * 2);   // packed
    __bf16* xbf  = (__bf16*)carve((size_t)NODES * FIN * 2);  // reused as hbf
    __bf16* eabf = (__bf16*)carve((size_t)EDGES * FEDGE * 2);
    float*  xl   = (float*)carve((size_t)NODES * HC * 4);
    float*  xr   = (float*)carve((size_t)NODES * HC * 4);
    float*  ebuf = (float*)carve((size_t)EDGES * HC * 4);
    float*  alph = (float*)carve((size_t)EDGES * HEADS * 4);
    unsigned* amax_u = (unsigned*)carve((size_t)NODES * HEADS * 4);
    float*  amax_f = (float*)carve((size_t)NODES * HEADS * 4);
    float*  denom  = (float*)carve((size_t)NODES * HEADS * 4);
    float*  acc    = (float*)carve((size_t)NODES * HC * 4);  // layer out accum

    const int TB = 256;
    auto blocks1d = [&](size_t n) { return (unsigned)((n + TB - 1) / TB); };

    // ---- weight packing + activation conversions ----
    pack_B_bf16<<<blocks1d((size_t)HC * HC), TB, 0, stream>>>(Wl1, wl1b, FIN, HC);
    pack_B_bf16<<<blocks1d((size_t)HC * HC), TB, 0, stream>>>(Wr1, wr1b, FIN, HC);
    pack_B_bf16<<<blocks1d((size_t)HC * HC), TB, 0, stream>>>(Wl2, wl2b, HC, HC);
    pack_B_bf16<<<blocks1d((size_t)HC * HC), TB, 0, stream>>>(Wr2, wr2b, HC, HC);
    pack_B_bf16<<<blocks1d((size_t)FEDGE * HC), TB, 0, stream>>>(We1, we1b, FEDGE, HC);
    pack_B_bf16<<<blocks1d((size_t)FEDGE * HC), TB, 0, stream>>>(We2, we2b, FEDGE, HC);
    cvt_f32_to_bf16<<<blocks1d((size_t)NODES * FIN), TB, 0, stream>>>(x, xbf, (size_t)NODES * FIN);
    cvt_f32_to_bf16<<<blocks1d((size_t)EDGES * FEDGE), TB, 0, stream>>>(ea, eabf, (size_t)EDGES * FEDGE);

    // grid sizes: one wave = one 16x64 strip; 8 waves per block
    const unsigned nodeJobs = ((NODES + 15) / 16) * (HC / 64);
    const unsigned nodeBlocks = (nodeJobs + 7) / 8;
    const unsigned edgeJobs = (EDGES / 16) * (HC / 64);
    const unsigned edgeBlocks = (edgeJobs + 7) / 8;
    const unsigned edgePairBlocks = (EDGES + 1) / 2;

    // ============================ LAYER 1 ==================================
    gemm_wmma_bf16<FIN><<<nodeBlocks, TB, 0, stream>>>(xbf, wl1b, bl1, xl, NODES, HC);
    gemm_wmma_bf16<FIN><<<nodeBlocks, TB, 0, stream>>>(xbf, wr1b, br1, xr, NODES, HC);
    gemm_wmma_bf16<FEDGE><<<edgeBlocks, TB, 0, stream>>>(eabf, we1b, nullptr, ebuf, EDGES, HC);

    fill_u32<<<blocks1d((size_t)NODES * HEADS), TB, 0, stream>>>(amax_u, ORD_NEG_INF, (size_t)NODES * HEADS);
    fill_f32<<<blocks1d((size_t)NODES * HEADS), TB, 0, stream>>>(denom, 0.f, (size_t)NODES * HEADS);
    fill_f32<<<blocks1d((size_t)NODES * HC), TB, 0, stream>>>(acc, 0.f, (size_t)NODES * HC);

    edge_alpha_kernel<<<edgePairBlocks, TB, 0, stream>>>(xl, xr, ebuf, srcI, dstI, att1, alph, amax_u);
    amax_fix_kernel<<<blocks1d((size_t)NODES * HEADS), TB, 0, stream>>>(amax_u, amax_f, (size_t)NODES * HEADS);
    edge_exp_kernel<<<blocks1d((size_t)EDGES * HEADS), TB, 0, stream>>>(alph, dstI, amax_f, denom);
    edge_aggr_kernel<<<edgePairBlocks, TB, 0, stream>>>(xl, srcI, dstI, alph, denom, acc);

    // h = ELU(out1 + bias1)  ->  bf16 (reusing xbf as hbf)
    node_finish1_kernel<<<blocks1d((size_t)NODES * HC), TB, 0, stream>>>(acc, bias1, xbf);

    // ============================ LAYER 2 ==================================
    gemm_wmma_bf16<HC><<<nodeBlocks, TB, 0, stream>>>(xbf, wl2b, bl2, xl, NODES, HC);
    gemm_wmma_bf16<HC><<<nodeBlocks, TB, 0, stream>>>(xbf, wr2b, br2, xr, NODES, HC);
    gemm_wmma_bf16<FEDGE><<<edgeBlocks, TB, 0, stream>>>(eabf, we2b, nullptr, ebuf, EDGES, HC);

    fill_u32<<<blocks1d((size_t)NODES * HEADS), TB, 0, stream>>>(amax_u, ORD_NEG_INF, (size_t)NODES * HEADS);
    fill_f32<<<blocks1d((size_t)NODES * HEADS), TB, 0, stream>>>(denom, 0.f, (size_t)NODES * HEADS);
    fill_f32<<<blocks1d((size_t)NODES * HC), TB, 0, stream>>>(acc, 0.f, (size_t)NODES * HC);

    edge_alpha_kernel<<<edgePairBlocks, TB, 0, stream>>>(xl, xr, ebuf, srcI, dstI, att2, alph, amax_u);
    amax_fix_kernel<<<blocks1d((size_t)NODES * HEADS), TB, 0, stream>>>(amax_u, amax_f, (size_t)NODES * HEADS);
    edge_exp_kernel<<<blocks1d((size_t)EDGES * HEADS), TB, 0, stream>>>(alph, dstI, amax_f, denom);
    edge_aggr_kernel<<<edgePairBlocks, TB, 0, stream>>>(xl, srcI, dstI, alph, denom, acc);

    // final: mean over heads + bias2 -> [NODES, 32]
    node_finish2_kernel<<<blocks1d((size_t)NODES * CH), TB, 0, stream>>>(acc, bias2, (float*)d_out);
}
